// GRUDModel_45148696215777
// MI455X (gfx1250) — matrix operand order repaired and checked by reading, not compiled, not run on hardware
//
#include <hip/hip_runtime.h>
#include <hip/hip_bf16.h>
#include <math.h>

// ---------------------------------------------------------------------------
// GRU-D forward, persistent-workgroup WMMA implementation for gfx1250.
// B=512 T=256 D=128 U=512.  32 workgroups x 512 threads (16 wave32).
// Each WG owns 16 batch rows (one WMMA M tile) and scans all T steps.
// Weights are pre-converted to bf16, N-major (N x K) in workspace.
// pre = x_t@K + (1-m)@M + bias is staged in LDS (fp32) to keep register
// pressure low (no scratch spills); z overwrites its pre slot in place.
// ---------------------------------------------------------------------------

#define GB 512    // batch
#define GT 256    // time
#define GD 128    // input dim
#define GU 512    // units
#define GU3 1536
#define PRE_LD 1544   // 1536 + 8 pad (floats)

typedef __bf16 bf16;
typedef bf16  bf16x8  __attribute__((ext_vector_type(8)));
typedef bf16  bf16x16 __attribute__((ext_vector_type(16)));
typedef float f32x8   __attribute__((ext_vector_type(8)));

#define SPLAT8(v) {v, v, v, v, v, v, v, v}

__device__ __forceinline__ f32x8 wmma_bf16(bf16x16 a, bf16x16 b, f32x8 c) {
  return __builtin_amdgcn_wmma_f32_16x16x32_bf16(false, a, false, b,
                                                 (short)0, c, false, false);
}

// A fragment (16x32 bf16) from LDS, row-major with stride ld (elements).
// Lanes 0-15: row=lane, K = k0+[0..7] and k0+[16..23]
// Lanes 16-31: row=lane-16, K = k0+[8..15] and k0+[24..31]
__device__ __forceinline__ bf16x16 ldsA(const bf16* m, int ld, int k0, int lane) {
  const int half = lane >> 4;
  const int r    = lane & 15;
  const bf16* p  = m + r * ld + k0 + half * 8;
  union { bf16x16 v; bf16x8 h[2]; } u;
  u.h[0] = *(const bf16x8*)(p);
  u.h[1] = *(const bf16x8*)(p + 16);
  return u.v;
}

// B fragment (32x16 bf16) from global, N-major weight (N rows x K cols).
// Lanes 0-15: N=n0+lane,   K = k0+[0..7] and k0+[16..23]
// Lanes 16-31: N=n0+lane-16, K = k0+[8..15] and k0+[24..31]
__device__ __forceinline__ bf16x16 glbB(const bf16* w, int ldk, int n0, int k0, int lane) {
  const int half = lane >> 4;
  const int nl   = lane & 15;
  const bf16* p  = w + (size_t)(n0 + nl) * ldk + k0 + half * 8;
  union { bf16x16 v; bf16x8 h[2]; } u;
  u.h[0] = *(const bf16x8*)(p);
  u.h[1] = *(const bf16x8*)(p + 16);
  return u.v;
}

__device__ __forceinline__ float sigmoidf_(float x) {
  return 1.0f / (1.0f + __expf(-x));
}

// ---------------------------------------------------------------------------
// one-time weight convert + transpose: fp32 (R x C) -> bf16 (C x R)
// ---------------------------------------------------------------------------
__global__ void cvt_transpose_kernel(const float* __restrict__ in,
                                     bf16* __restrict__ out, int R, int C) {
  int i = blockIdx.x * blockDim.x + threadIdx.x;
  if (i >= R * C) return;
  int r = i / C, c = i - r * C;
  out[(size_t)c * R + r] = (bf16)in[i];
}

// ---------------------------------------------------------------------------
// main persistent kernel
// ---------------------------------------------------------------------------
__global__ void __launch_bounds__(512, 1)
grud_persistent_kernel(const float* __restrict__ values,   // (B,T,D)
                       const int*   __restrict__ maskp,    // (B,T,D)
                       const float* __restrict__ times,    // (B,T,1)
                       const float* __restrict__ bias,     // (3U)
                       const float* __restrict__ idk,      // (D)
                       const float* __restrict__ idb,      // (D)
                       const float* __restrict__ hdb,      // (U)
                       float*       __restrict__ out,      // (B,U)
                       const bf16*  __restrict__ WkT,      // (3U x D)
                       const bf16*  __restrict__ WmT,      // (3U x D)
                       const bf16*  __restrict__ WhdT,     // (U x D)
                       const bf16*  __restrict__ RT)       // (3U x U): rows 0..1023 = Rzr cols, 1024..1535 = Rh cols
{
  __shared__ __align__(16) bf16  s_xt[16][136];
  __shared__ __align__(16) bf16  s_om[16][136];
  __shared__ __align__(16) bf16  s_dd[16][136];
  __shared__ __align__(16) bf16  s_hd[16][520];
  __shared__ __align__(16) bf16  s_rhd[16][520];
  __shared__ __align__(16) float s_h[16][516];
  __shared__ __align__(16) float s_pre[16][PRE_LD];  // pre; cols [0,512) become z in phase C
  __shared__ __align__(16) float s_xk[16][132];
  __shared__ __align__(16) float s_sp[16][132];

  const int tid   = threadIdx.x;
  const int wave  = tid >> 5;
  const int lane  = tid & 31;
  const int bbase = blockIdx.x * 16;

  // ---- init carries -------------------------------------------------------
  for (int i = tid; i < 16 * 512; i += 512) {
    s_h[i >> 9][i & 511] = 0.0f;
  }
  for (int i = tid; i < 16 * 128; i += 512) {
    int r = i >> 7, c = i & 127;
    s_xk[r][c] = 0.0f;
    s_sp[r][c] = times[(size_t)(bbase + r) * GT + 0];
  }
  __syncthreads();

  const int nlane = lane & 15;
  const int rb    = (lane >> 4) * 8;

  for (int t = 0; t < GT; ++t) {
    // ---- Phase A: elementwise input/decay prep (fp32) ---------------------
    for (int i = tid; i < 16 * 128; i += 512) {
      int r = i >> 7, c = i & 127;
      size_t gi = ((size_t)(bbase + r) * GT + t) * GD + c;
      float x  = values[gi];
      int   mi = maskp[gi];
      float s  = times[(size_t)(bbase + r) * GT + t];
      float sp = s_sp[r][c];
      float d  = s - sp;
      float gdi = __expf(-fmaxf(d * idk[c] + idb[c], 0.0f));
      float xk  = s_xk[r][c];
      float xt, om;
      if (mi) { xk = x; xt = x; om = 0.0f; s_sp[r][c] = s; }
      else    { xt = gdi * xk; om = 1.0f; }
      s_xk[r][c] = xk;
      s_xt[r][c] = (bf16)xt;
      s_om[r][c] = (bf16)om;
      s_dd[r][c] = (bf16)d;
    }
    __syncthreads();

    // ---- Phase B1: gamma_dh pair -> h_d (bf16 in LDS) ---------------------
    {
      const int n0a = (wave * 2) * 16;
      const int n0b = n0a + 16;
      f32x8 acc0 = SPLAT8(0.0f);
      f32x8 acc1 = SPLAT8(0.0f);
      #pragma unroll 2
      for (int k0 = 0; k0 < GD; k0 += 32) {
        bf16x16 av = ldsA(&s_dd[0][0], 136, k0, lane);
        acc0 = wmma_bf16(av, glbB(WhdT, GD, n0a, k0, lane), acc0);
        acc1 = wmma_bf16(av, glbB(WhdT, GD, n0b, k0, lane), acc1);
      }
      int na = n0a + nlane, nb = n0b + nlane;
      float ba = hdb[na], bb = hdb[nb];
      #pragma unroll
      for (int j = 0; j < 8; ++j) {
        float g0 = __expf(-fmaxf(acc0[j] + ba, 0.0f));
        float g1 = __expf(-fmaxf(acc1[j] + bb, 0.0f));
        s_hd[rb + j][na] = (bf16)(g0 * s_h[rb + j][na]);
        s_hd[rb + j][nb] = (bf16)(g1 * s_h[rb + j][nb]);
      }
    }

    // ---- Phase B2: pre = bias + x_t@K + (1-m)@M, 6 tiles/wave -> LDS ------
    #pragma unroll
    for (int pp = 0; pp < 3; ++pp) {
      const int n0a = (wave * 6 + pp * 2) * 16;
      const int n0b = n0a + 16;
      float bva = bias[n0a + nlane];
      float bvb = bias[n0b + nlane];
      f32x8 acc0 = SPLAT8(bva);
      f32x8 acc1 = SPLAT8(bvb);
      #pragma unroll 2
      for (int k0 = 0; k0 < GD; k0 += 32) {
        bf16x16 ax = ldsA(&s_xt[0][0], 136, k0, lane);
        bf16x16 ao = ldsA(&s_om[0][0], 136, k0, lane);
        acc0 = wmma_bf16(ax, glbB(WkT, GD, n0a, k0, lane), acc0);
        acc0 = wmma_bf16(ao, glbB(WmT, GD, n0a, k0, lane), acc0);
        acc1 = wmma_bf16(ax, glbB(WkT, GD, n0b, k0, lane), acc1);
        acc1 = wmma_bf16(ao, glbB(WmT, GD, n0b, k0, lane), acc1);
      }
      int na = n0a + nlane, nb = n0b + nlane;
      #pragma unroll
      for (int j = 0; j < 8; ++j) {
        s_pre[rb + j][na] = acc0[j];
        s_pre[rb + j][nb] = acc1[j];
      }
    }
    __syncthreads();   // s_hd, s_pre complete

    // ---- Phase C: pre_zr += h_d @ Rzr; z in place; rhd = r * h_d ----------
    #pragma unroll
    for (int pp = 0; pp < 2; ++pp) {
      const int n0a = (wave * 4 + pp * 2) * 16;   // zr column tiles [0,1024)
      const int n0b = n0a + 16;
      f32x8 acc0 = SPLAT8(0.0f);
      f32x8 acc1 = SPLAT8(0.0f);
      #pragma unroll 2
      for (int k0 = 0; k0 < GU; k0 += 32) {
        bf16x16 av = ldsA(&s_hd[0][0], 520, k0, lane);
        acc0 = wmma_bf16(av, glbB(RT, GU, n0a, k0, lane), acc0);
        acc1 = wmma_bf16(av, glbB(RT, GU, n0b, k0, lane), acc1);
      }
      int na = n0a + nlane, nb = n0b + nlane;
      #pragma unroll
      for (int j = 0; j < 8; ++j) {
        float v0 = sigmoidf_(s_pre[rb + j][na] + acc0[j]);
        float v1 = sigmoidf_(s_pre[rb + j][nb] + acc1[j]);
        if (na < GU) {
          s_pre[rb + j][na] = v0;                        // z gate (in place)
        } else {
          int u = na - GU;                               // r gate
          s_rhd[rb + j][u] = (bf16)(v0 * (float)s_hd[rb + j][u]);
        }
        if (nb < GU) {
          s_pre[rb + j][nb] = v1;
        } else {
          int u = nb - GU;
          s_rhd[rb + j][u] = (bf16)(v1 * (float)s_hd[rb + j][u]);
        }
      }
    }
    __syncthreads();   // s_rhd, z complete

    // ---- Phase D: hh = tanh(pre_h + rhd @ Rh); h = z*h + (1-z)*hh ---------
    {
      const int n0a = (wave * 2) * 16;            // unit tiles [0,512)
      const int n0b = n0a + 16;
      f32x8 acc0 = SPLAT8(0.0f);
      f32x8 acc1 = SPLAT8(0.0f);
      #pragma unroll 2
      for (int k0 = 0; k0 < GU; k0 += 32) {
        bf16x16 av = ldsA(&s_rhd[0][0], 520, k0, lane);
        acc0 = wmma_bf16(av, glbB(RT, GU, 2 * GU + n0a, k0, lane), acc0);
        acc1 = wmma_bf16(av, glbB(RT, GU, 2 * GU + n0b, k0, lane), acc1);
      }
      int na = n0a + nlane, nb = n0b + nlane;
      #pragma unroll
      for (int j = 0; j < 8; ++j) {
        float hh0 = tanhf(s_pre[rb + j][2 * GU + na] + acc0[j]);
        float hh1 = tanhf(s_pre[rb + j][2 * GU + nb] + acc1[j]);
        float z0  = s_pre[rb + j][na];              // z stored in place
        float z1  = s_pre[rb + j][nb];
        s_h[rb + j][na] = z0 * s_h[rb + j][na] + (1.0f - z0) * hh0;
        s_h[rb + j][nb] = z1 * s_h[rb + j][nb] + (1.0f - z1) * hh1;
      }
    }
    __syncthreads();   // next step reads s_h / overwrites staging
  }

  // ---- write final h ------------------------------------------------------
  for (int i = tid; i < 16 * 512; i += 512) {
    int r = i >> 9, c = i & 511;
    out[(size_t)(bbase + r) * GU + c] = s_h[r][c];
  }
}

// ---------------------------------------------------------------------------
// host entry
// ---------------------------------------------------------------------------
extern "C" void kernel_launch(void* const* d_in, const int* in_sizes, int n_in,
                              void* d_out, int out_size, void* d_ws, size_t ws_size,
                              hipStream_t stream) {
  const float* values  = (const float*)d_in[0];
  const int*   mask    = (const int*)  d_in[1];
  const float* times   = (const float*)d_in[2];
  const float* kern    = (const float*)d_in[3];   // (D, 3U)
  const float* rkern   = (const float*)d_in[4];   // (U, 3U)
  const float* bias    = (const float*)d_in[5];   // (3U)
  const float* idk     = (const float*)d_in[6];   // (D)
  const float* idb     = (const float*)d_in[7];   // (D)
  const float* hdk     = (const float*)d_in[8];   // (D, U)
  const float* hdb     = (const float*)d_in[9];   // (U)
  const float* mkern   = (const float*)d_in[10];  // (D, 3U)
  float* out = (float*)d_out;

  // workspace layout (bf16 elements)
  bf16* w     = (bf16*)d_ws;
  bf16* WkT   = w;                         // 1536 x 128
  bf16* WmT   = w + (size_t)GU3 * GD;      // 1536 x 128
  bf16* WhdT  = WmT + (size_t)GU3 * GD;    // 512 x 128
  bf16* RT    = WhdT + (size_t)GU * GD;    // 1536 x 512

  {
    int n = GD * GU3;
    cvt_transpose_kernel<<<(n + 255) / 256, 256, 0, stream>>>(kern,  WkT,  GD, GU3);
    cvt_transpose_kernel<<<(n + 255) / 256, 256, 0, stream>>>(mkern, WmT,  GD, GU3);
    int n2 = GD * GU;
    cvt_transpose_kernel<<<(n2 + 255) / 256, 256, 0, stream>>>(hdk, WhdT, GD, GU);
    int n3 = GU * GU3;
    cvt_transpose_kernel<<<(n3 + 255) / 256, 256, 0, stream>>>(rkern, RT, GU, GU3);
  }

  grud_persistent_kernel<<<dim3(GB / 16), dim3(512), 0, stream>>>(
      values, mask, times, bias, idk, idb, hdb, out, WkT, WmT, WhdT, RT);
}
